// MoCA_27908697490123
// MI455X (gfx1250) — matrix operand order repaired and checked by reading, not compiled, not run on hardware
//
#include <hip/hip_runtime.h>

// MoCA pipeline for MI455X (gfx1250, wave32, WMMA).
// B=8, S=2048, D=1024. GEMMs (~206 GFLOP, dominant) run in bf16 with f32
// accumulation via v_wmma_f32_16x16x32_bf16. All elementwise/conv kernels are
// bandwidth-bound and fully coalesced; the transpose goes through an LDS tile.

typedef __attribute__((ext_vector_type(16))) __bf16         v16bf;
typedef __attribute__((ext_vector_type(8)))  float          v8f;
typedef __attribute__((ext_vector_type(8)))  unsigned short u16x8;

#define BB  8
#define SSZ 2048
#define DSZ 1024

static constexpr size_t nBSD = (size_t)BB * SSZ * DSZ;   // 16,777,216
static constexpr size_t nBSS = (size_t)BB * SSZ * SSZ;   // 33,554,432

// ---------------------------------------------------------------------------
// WMMA fragment loaders (CDNA5 ISA §7.12.2).
// ---------------------------------------------------------------------------
union FragU { v16bf v; u16x8 h[2]; };

// A: 16x32 tile of a row-major [M,K] matrix, lda = K.
// Lanes 0-15: M=lane, elems 0..7 = K 0..7,  elems 8..15 = K 16..23.
// Lanes 16-31: M=lane-16, elems 0..7 = K 8..15, elems 8..15 = K 24..31.
__device__ __forceinline__ v16bf load_a_frag(const __bf16* tile, int lda, int lane) {
    int m  = lane & 15;
    int kb = (lane >> 4) << 3;   // 0 or 8
    const __bf16* p = tile + (size_t)m * lda + kb;
    FragU u;
    u.h[0] = *(const u16x8*)(p);
    u.h[1] = *(const u16x8*)(p + 16);
    return u.v;
}

// B: 32x16 tile of a row-major [K,N] matrix, ldb = N. N = lane%16;
// lanes 0-15: elems 0..7 = K 0..7, 8..15 = K 16..23; lanes 16-31: K 8..15, 24..31.
__device__ __forceinline__ v16bf load_b_frag(const __bf16* tile, int ldb, int lane) {
    int n  = lane & 15;
    int kb = (lane >> 4) << 3;
    v16bf r;
#pragma unroll
    for (int s = 0; s < 8; ++s) {
        r[s]     = tile[(size_t)(kb + s) * ldb + n];
        r[s + 8] = tile[(size_t)(16 + kb + s) * ldb + n];
    }
    return r;
}

// ---------------------------------------------------------------------------
// GEMM: C[z] = A[z] (MxK, bf16) * B[z] (KxN, bf16) -> f32, optional +residual
// (compile-time flag -> no branches in the epilogue).
// Block = 256 thr = 8 waves (2x4); wave tile 32x64; block tile 64x256.
// Grid: (M/64, N/256, batch). No divergence before WMMA (EXEC all 1s).
// ---------------------------------------------------------------------------
template <bool HAS_RES>
__global__ __launch_bounds__(256) void gemm_bf16_nn_kernel(
    const __bf16* __restrict__ A, const __bf16* __restrict__ B,
    float* __restrict__ C, const float* __restrict__ residual,
    int M, int N, int K)
{
    const int lane = threadIdx.x & 31;
    const int wave = threadIdx.x >> 5;
    const int row0 = blockIdx.x * 64  + (wave >> 2) * 32;
    const int col0 = blockIdx.y * 256 + (wave & 3) * 64;

    const __bf16* Ab = A + (size_t)blockIdx.z * M * K;
    const __bf16* Bb = B + (size_t)blockIdx.z * K * N;
    const size_t  zC = (size_t)blockIdx.z * M * N;

    v8f acc[2][4] = {};

    for (int k = 0; k < K; k += 32) {
        if (k + 32 < K) {  // uniform branch: prefetch next K tile into GL2
            __builtin_prefetch(Ab + (size_t)row0 * K + (k + 32), 0, 0);
            __builtin_prefetch(Bb + (size_t)(k + 32) * N + col0, 0, 0);
        }
        v16bf a[2];
        a[0] = load_a_frag(Ab + (size_t)(row0)      * K + k, K, lane);
        a[1] = load_a_frag(Ab + (size_t)(row0 + 16) * K + k, K, lane);
        v16bf b[4];
#pragma unroll
        for (int ni = 0; ni < 4; ++ni)
            b[ni] = load_b_frag(Bb + (size_t)k * N + col0 + ni * 16, N, lane);
#pragma unroll
        for (int mi = 0; mi < 2; ++mi)
#pragma unroll
            for (int ni = 0; ni < 4; ++ni)
                acc[mi][ni] = __builtin_amdgcn_wmma_f32_16x16x32_bf16(
                    false, a[mi], false, b[ni], (short)0, acc[mi][ni],
                    false, false);
    }

    // C/D layout: lane 0-15 -> N=lane, M = vgpr r; lane 16-31 -> M = 8 + r.
    const int nn = lane & 15;
    const int mo = (lane >> 4) << 3;
#pragma unroll
    for (int mi = 0; mi < 2; ++mi)
#pragma unroll
        for (int ni = 0; ni < 4; ++ni)
#pragma unroll
            for (int r = 0; r < 8; ++r) {
                size_t off = zC + (size_t)(row0 + mi * 16 + mo + r) * N
                                + (size_t)(col0 + ni * 16 + nn);
                float v = acc[mi][ni][r];
                if constexpr (HAS_RES) v += residual[off];
                C[off] = v;
            }
}

// ---------------------------------------------------------------------------
// LDS-tiled 32x32 transpose of x: emits
//   xb  = bf16(x)        [B,S,D]   (coalesced)
//   xTf = f32 transpose  [B,D,S]   (coalesced, feeds the conv on xT)
//   xbT = bf16 transpose [B,D,S]   (coalesced, GEMM B operand)
// Grid: (D/32, S/32, B), block 256 = 32 lanes x 8 rows.
// ---------------------------------------------------------------------------
__global__ __launch_bounds__(256) void transpose_cast_kernel(
    const float* __restrict__ x, __bf16* __restrict__ xb,
    float* __restrict__ xTf, __bf16* __restrict__ xbT)
{
    __shared__ float tile[32][33];
    const int tx = threadIdx.x & 31;
    const int ty = threadIdx.x >> 5;       // 0..7
    const int d0 = blockIdx.x * 32;
    const int s0 = blockIdx.y * 32;
    const int b  = blockIdx.z;

    const float* xbase = x + (size_t)b * SSZ * DSZ;
#pragma unroll
    for (int r = 0; r < 4; ++r) {
        int sl = ty * 4 + r;
        int s  = s0 + sl;
        float v = xbase[(size_t)s * DSZ + d0 + tx];
        tile[sl][tx] = v;
        xb[((size_t)b * SSZ + s) * DSZ + d0 + tx] = (__bf16)v;
    }
    __syncthreads();

    float*  xT_base  = xTf + (size_t)b * DSZ * SSZ;
    __bf16* xbT_base = xbT + (size_t)b * DSZ * SSZ;
#pragma unroll
    for (int r = 0; r < 4; ++r) {
        int dl = ty * 4 + r;
        int d  = d0 + dl;
        float v = tile[tx][dl];
        size_t o = (size_t)d * SSZ + s0 + tx;
        xT_base[o]  = v;
        xbT_base[o] = (__bf16)v;
    }
}

// ---------------------------------------------------------------------------
// Generic 1-channel 5x5 SAME conv on row-major [B,1,H,W] f32, bf16 out.
// All reads/writes coalesced (adjacent threads walk W).
// ---------------------------------------------------------------------------
__global__ __launch_bounds__(256) void conv5x5_kernel(
    const float* __restrict__ in, const float* __restrict__ w,
    const float* __restrict__ bias, __bf16* __restrict__ out, int H, int W)
{
    size_t idx = (size_t)blockIdx.x * 256 + threadIdx.x;   // < B*H*W
    int wc = (int)(idx % (size_t)W);
    size_t t = idx / (size_t)W;
    int h = (int)(t % (size_t)H);
    int b = (int)(t / (size_t)H);
    float wr[25];
#pragma unroll
    for (int q = 0; q < 25; ++q) wr[q] = w[q];
    float acc = bias[0];
#pragma unroll
    for (int i = 0; i < 5; ++i) {
        int hh = h + i - 2;
        if ((unsigned)hh >= (unsigned)H) continue;
        const float* row = in + ((size_t)b * H + hh) * (size_t)W;
#pragma unroll
        for (int j = 0; j < 5; ++j) {
            int ww = wc + j - 2;
            if ((unsigned)ww < (unsigned)W) acc += wr[i * 5 + j] * row[ww];
        }
    }
    out[idx] = (__bf16)acc;
}

// 2-channel 5x5 SAME conv on [B,2,S,S] (NSSM, Attention) -> bf16 Moca map.
__global__ __launch_bounds__(256) void conv5x5_moca_kernel(
    const float* __restrict__ p0, const float* __restrict__ p1,
    const float* __restrict__ w, const float* __restrict__ bias,
    __bf16* __restrict__ out)
{
    size_t idx = (size_t)blockIdx.x * 256 + threadIdx.x;   // < B*S*S
    int wc = (int)(idx % SSZ);
    size_t t = idx / SSZ;
    int h = (int)(t % SSZ);
    int b = (int)(t / SSZ);
    float wr[50];
#pragma unroll
    for (int q = 0; q < 50; ++q) wr[q] = w[q];
    float acc = bias[0];
#pragma unroll
    for (int i = 0; i < 5; ++i) {
        int hh = h + i - 2;
        if ((unsigned)hh >= SSZ) continue;
        size_t base = ((size_t)b * SSZ + hh) * SSZ;
#pragma unroll
        for (int j = 0; j < 5; ++j) {
            int ww = wc + j - 2;
            if ((unsigned)ww < SSZ) {
                acc += wr[i * 5 + j]      * p0[base + ww];
                acc += wr[25 + i * 5 + j] * p1[base + ww];
            }
        }
    }
    out[idx] = (__bf16)acc;
}

// Softmax over the BATCH axis (axis=0, per the reference), in place on a
// [B,S,S] f32 map. One thread per (s,t), 8 plane-strided values.
__global__ __launch_bounds__(256) void softmax_batch_kernel(float* __restrict__ m)
{
    size_t idx = (size_t)blockIdx.x * 256 + threadIdx.x;   // < S*S
    constexpr size_t plane = (size_t)SSZ * SSZ;
    float v[BB];
    float mx = -3.4e38f;
#pragma unroll
    for (int b = 0; b < BB; ++b) {
        v[b] = m[(size_t)b * plane + idx];
        mx = fmaxf(mx, v[b]);
    }
    float sum = 0.f;
#pragma unroll
    for (int b = 0; b < BB; ++b) {
        v[b] = __expf(v[b] - mx);
        sum += v[b];
    }
    float inv = 1.f / sum;
#pragma unroll
    for (int b = 0; b < BB; ++b)
        m[(size_t)b * plane + idx] = v[b] * inv;
}

// ---------------------------------------------------------------------------
extern "C" void kernel_launch(void* const* d_in, const int* in_sizes, int n_in,
                              void* d_out, int out_size, void* d_ws, size_t ws_size,
                              hipStream_t stream)
{
    const float* x  = (const float*)d_in[0];
    const float* w1 = (const float*)d_in[1];
    const float* b1 = (const float*)d_in[2];
    const float* w2 = (const float*)d_in[3];
    const float* b2 = (const float*)d_in[4];
    const float* w3 = (const float*)d_in[5];
    const float* b3 = (const float*)d_in[6];
    const float* w4 = (const float*)d_in[7];
    const float* b4 = (const float*)d_in[8];
    float* out = (float*)d_out;

    // Workspace layout (544 MB total).
    __bf16* xb    = (__bf16*)d_ws;          // bf16 x        [B,S,D]  32 MB
    __bf16* xbT   = xb    + nBSD;           // bf16 x^T      [B,D,S]  32 MB
    __bf16* cxb   = xbT   + nBSD;           // conv_x  bf16  [B,S,D]  32 MB
    __bf16* cxtb  = cxb   + nBSD;           // conv_xT bf16  [B,D,S]  32 MB
    __bf16* cgb   = cxtb  + nBSD;           // conv_g  bf16  [B,S,D]  32 MB
    float*  xTf   = (float*)(cgb + nBSD);   // f32 x^T       [B,D,S]  64 MB
    float*  sc1   = xTf   + nBSD;           // NSSM map f32  [B,S,S] 128 MB
    float*  sc2   = sc1   + nBSS;           // Attn map f32  [B,S,S] 128 MB
    __bf16* mocab = (__bf16*)(sc2 + nBSS);  // Moca map bf16 [B,S,S]  64 MB

    const int ELT_BSD = (int)(nBSD / 256);  // 65536
    const int ELT_BSS = (int)(nBSS / 256);  // 131072
    const int ELT_SS  = (SSZ * SSZ) / 256;  // 16384

    // 1) bf16 cast + LDS-tiled transpose (f32 and bf16 copies of x^T).
    {
        dim3 grid(DSZ / 32, SSZ / 32, BB);
        transpose_cast_kernel<<<grid, 256, 0, stream>>>(x, xb, xTf, xbT);
    }

    // 2) convs feeding the GEMMs (bf16 outputs, all coalesced).
    conv5x5_kernel<<<ELT_BSD, 256, 0, stream>>>(x,   w1, b1, cxb,  SSZ, DSZ);
    conv5x5_kernel<<<ELT_BSD, 256, 0, stream>>>(xTf, w2, b2, cxtb, DSZ, SSZ);
    conv5x5_kernel<<<ELT_BSD, 256, 0, stream>>>(x,   w4, b4, cgb,  SSZ, DSZ);

    // 3) scores1 = x @ x^T   (M=S, N=S, K=D)
    {
        dim3 grid(SSZ / 64, SSZ / 256, BB);
        gemm_bf16_nn_kernel<false><<<grid, 256, 0, stream>>>(
            xb, xbT, sc1, nullptr, SSZ, SSZ, DSZ);
    }
    // 4) scores2 = conv_x @ conv_xT
    {
        dim3 grid(SSZ / 64, SSZ / 256, BB);
        gemm_bf16_nn_kernel<false><<<grid, 256, 0, stream>>>(
            cxb, cxtb, sc2, nullptr, SSZ, SSZ, DSZ);
    }

    // 5) batch-axis softmax, in place.
    softmax_batch_kernel<<<ELT_SS, 256, 0, stream>>>(sc1);
    softmax_batch_kernel<<<ELT_SS, 256, 0, stream>>>(sc2);

    // 6) Moca map = conv5x5([NSSM, Attention]) -> bf16.
    conv5x5_moca_kernel<<<ELT_BSS, 256, 0, stream>>>(sc1, sc2, w3, b3, mocab);

    // 7) out = Moca @ conv_g + x   (M=S, N=D, K=S), residual fused.
    {
        dim3 grid(SSZ / 64, DSZ / 256, BB);
        gemm_bf16_nn_kernel<true><<<grid, 256, 0, stream>>>(
            mocab, cgb, out, x, SSZ, DSZ, SSZ);
    }
}
